// Structure2Vec_88399016886796
// MI455X (gfx1250) — matrix operand order repaired and checked by reading, not compiled, not run on hardware
//
#include <hip/hip_runtime.h>

// ---------------------------------------------------------------------------
// Structure2Vec forward for MI455X (gfx1250, wave32, WMMA).
// - GEMMs: v_wmma_f32_16x16x32_bf16 (bf16 operands, fp32 accumulate).
// - Edge bond-GEMM fused with segment_sum scatter (fp32 atomics, L2-resident)
//   and with the feat[src] gather/scatter; 64 edges/block so the two
//   B fragments are loaded once and reused across 4 WMMA sub-tiles.
// - Node GEMM: 64-row x 256-col block tile, 32x64 per wave (8 WMMA/K-step);
//   A fragments loaded as float4 (global_load_b128) runs.
// - BatchNorm stats reduced in GEMM epilogue (LDS ds_add_f32 + one global
//   atomic per column per block); BN-apply folded into next GEMM's A-load.
// - All hot-path offsets are 32-bit so the backend can use SGPR-base +
//   VGPR-offset (GVS) addressing instead of 64-bit VALU address math.
// ---------------------------------------------------------------------------

typedef __attribute__((ext_vector_type(16))) __bf16 bf16x16;
typedef __attribute__((ext_vector_type(8)))  float  floatx8;

constexpr int HDIM = 256;   // hidden size
constexpr int BDIM = 32;    // bond feature size
constexpr int ETILE = 64;   // edges per block in edge kernel
constexpr float BN_EPS = 1e-5f;

// pack 4 floats (optionally BN-transformed) into af[base..base+3] as bf16
__device__ __forceinline__ void pack4(bf16x16& af, int base, float4 p,
                                      float4 s, float4 h, bool bn) {
    if (bn) {
        af[base + 0] = (__bf16)(p.x * s.x + h.x);
        af[base + 1] = (__bf16)(p.y * s.y + h.y);
        af[base + 2] = (__bf16)(p.z * s.z + h.z);
        af[base + 3] = (__bf16)(p.w * s.w + h.w);
    } else {
        af[base + 0] = (__bf16)p.x;
        af[base + 1] = (__bf16)p.y;
        af[base + 2] = (__bf16)p.z;
        af[base + 3] = (__bf16)p.w;
    }
}

// ---------------------------------------------------------------- zero ------
__global__ __launch_bounds__(256)
void zero_kernel(float* __restrict__ p, long n) {
    long i = ((long)blockIdx.x * 256 + threadIdx.x) * 4;
    if (i + 3 < n) {
        *(float4*)(p + i) = make_float4(0.f, 0.f, 0.f, 0.f);
    } else {
        for (long j = i; j < n; ++j) p[j] = 0.f;
    }
}

// ------------------------------------------------------------ edge fused ----
__global__ __launch_bounds__(256)
void edge_kernel(const float* __restrict__ w,      // [E, 32]
                 const float* __restrict__ bW,     // [32, 256]
                 const float* __restrict__ bb,     // [256]
                 const int*   __restrict__ dst,
                 const int*   __restrict__ src,    // null if !do_gather
                 const float* __restrict__ feat,   // null if !do_gather
                 float* __restrict__ acc1,         // null if !do_gather
                 float* __restrict__ acc2,
                 int E, int do_gather) {
    const int tid  = threadIdx.x;
    const int lane = tid & 31;
    const int wv   = tid >> 5;                  // 0..7
    const int e0   = blockIdx.x * ETILE;
    const bool full = (e0 + ETILE) <= E;        // uniform

    __shared__ int s_dst[ETILE];
    __shared__ int s_src[ETILE];
    if (tid < ETILE) {
        int e = e0 + tid;
        s_dst[tid] = (e < E) ? dst[e] : -1;
        s_src[tid] = (do_gather && e < E) ? src[e] : -1;
    }
    __syncthreads();

    const int r  = lane & 15;
    const int hk = lane >> 4;
    const int colBase = wv * 32;

    // B fragments for this wave's 32 columns: loaded once, reused 4x.
    bf16x16 bf[2];
    #pragma unroll
    for (int a = 0; a < 2; ++a) {
        const unsigned col = colBase + a * 16 + r;
        #pragma unroll
        for (int j = 0; j < 16; ++j)
            bf[a][j] = (__bf16)bW[(unsigned)((j < 8 ? 0 : 8) + hk * 8 + j) * HDIM + col];
    }
    const float bias0 = bb[colBase + r];
    const float bias1 = bb[colBase + 16 + r];
    const unsigned col0 = colBase + r;
    const unsigned col1 = colBase + 16 + r;

    #pragma unroll
    for (int t = 0; t < 4; ++t) {               // 4 sub-tiles of 16 edges
        const int e = e0 + t * 16 + r;
        bf16x16 af;
        if (full || e < E) {
            const float4* wr4 = (const float4*)w + (unsigned)e * (BDIM / 4);
            float4 p0 = wr4[hk * 2 + 0];        // k = hk*8 + 0..3
            float4 p1 = wr4[hk * 2 + 1];        // k = hk*8 + 4..7
            float4 p2 = wr4[hk * 2 + 4];        // k = 16 + hk*8 + 0..3
            float4 p3 = wr4[hk * 2 + 5];        // k = 16 + hk*8 + 4..7
            float4 z = {};
            pack4(af, 0,  p0, z, z, false);
            pack4(af, 4,  p1, z, z, false);
            pack4(af, 8,  p2, z, z, false);
            pack4(af, 12, p3, z, z, false);
        } else {
            #pragma unroll
            for (int j = 0; j < 16; ++j) af[j] = (__bf16)0.0f;
        }

        floatx8 c0 = {}, c1 = {};
        c0 = __builtin_amdgcn_wmma_f32_16x16x32_bf16(false, af, false, bf[0],
                                                     (short)0, c0, false, false);
        c1 = __builtin_amdgcn_wmma_f32_16x16x32_bf16(false, af, false, bf[1],
                                                     (short)0, c1, false, false);

        if (full) {
            #pragma unroll
            for (int i = 0; i < 8; ++i) {
                const unsigned d = (unsigned)s_dst[t * 16 + i + hk * 8];
                const unsigned off = d * HDIM;
                atomicAdd(acc2 + (off + col0), c0[i] + bias0);
                atomicAdd(acc2 + (off + col1), c1[i] + bias1);
            }
        } else {
            #pragma unroll
            for (int i = 0; i < 8; ++i) {
                const int d = s_dst[t * 16 + i + hk * 8];
                if (d >= 0) {
                    const unsigned off = (unsigned)d * HDIM;
                    atomicAdd(acc2 + (off + col0), c0[i] + bias0);
                    atomicAdd(acc2 + (off + col1), c1[i] + bias1);
                }
            }
        }
    }

    // neighbor gather/scatter: 256 threads sweep one 1KB feat row per edge
    if (do_gather) {
        if (full) {
            #pragma unroll 4
            for (int e = 0; e < ETILE; ++e) {
                const unsigned doff = (unsigned)s_dst[e] * HDIM + tid;
                const unsigned soff = (unsigned)s_src[e] * HDIM + tid;
                atomicAdd(acc1 + doff, feat[soff]);
            }
        } else {
            for (int e = 0; e < ETILE; ++e) {
                const int d = s_dst[e];
                if (d >= 0) {
                    const unsigned doff = (unsigned)d * HDIM + tid;
                    const unsigned soff = (unsigned)s_src[e] * HDIM + tid;
                    atomicAdd(acc1 + doff, feat[soff]);
                }
            }
        }
    }
}

// ------------------------------------------------------------- node GEMM ----
// out = relu(A' @ W + bias + addv), A' = scaleA ? (A*scaleA+shiftA) : A.
// Block = 256 threads = 8 waves; block tile = 64 rows x 256 cols; each wave
// computes a 32x64 tile (2 A fragments x 4 B fragments = 8 WMMA / K-step).
// Column sums / sums-of-squares of out accumulate into stats[0..511].
__global__ __launch_bounds__(256)
void node_gemm_kernel(const float* __restrict__ A, int K,
                      const float* __restrict__ scaleA,   // [K] or null
                      const float* __restrict__ shiftA,   // [K] or null
                      const float* __restrict__ W,        // [K, 256]
                      const float* __restrict__ bias,     // [256]
                      const float* __restrict__ addv,     // [N, 256] or null
                      float* __restrict__ out,            // [N, 256]
                      float* __restrict__ stats,          // [512] sum|sumsq
                      int N) {
    __shared__ float s_sum[HDIM];
    __shared__ float s_ssq[HDIM];
    const int tid = threadIdx.x;
    if (tid < HDIM) { s_sum[tid] = 0.f; s_ssq[tid] = 0.f; }
    __syncthreads();

    const int lane = tid & 31;
    const int wv   = tid >> 5;
    const int r    = lane & 15;
    const int hk   = lane >> 4;
    const int rowBase = blockIdx.x * 64 + (wv >> 2) * 32;
    const int colBase = (wv & 3) * 64;
    const bool blockFull = (blockIdx.x + 1) * 64 <= N;   // uniform
    const bool bn = (scaleA != nullptr);

    floatx8 acc[2][4] = {};
    const int row0 = rowBase + r;
    const int row1 = rowBase + 16 + r;
    const unsigned aoff[2] = { (unsigned)row0 * (unsigned)K,
                               (unsigned)row1 * (unsigned)K };
    const bool rok[2] = { blockFull || row0 < N, blockFull || row1 < N };

    for (int kb = 0; kb < K; kb += 32) {
        bf16x16 af[2];
        float4 s0, s1, s2, s3, h0, h1, h2, h3;
        if (bn) {
            const float4* sc4 = (const float4*)(scaleA + kb);
            const float4* sh4 = (const float4*)(shiftA + kb);
            const int b = hk * 2;
            s0 = sc4[b]; s1 = sc4[b + 1]; s2 = sc4[b + 4]; s3 = sc4[b + 5];
            h0 = sh4[b]; h1 = sh4[b + 1]; h2 = sh4[b + 4]; h3 = sh4[b + 5];
        } else {
            s0 = s1 = s2 = s3 = make_float4(0, 0, 0, 0);
            h0 = h1 = h2 = h3 = make_float4(0, 0, 0, 0);
        }
        #pragma unroll
        for (int g = 0; g < 2; ++g) {
            if (rok[g]) {
                const float4* a4 = (const float4*)A + (aoff[g] + kb) / 4;
                const int b = hk * 2;
                float4 p0 = a4[b], p1 = a4[b + 1], p2 = a4[b + 4], p3 = a4[b + 5];
                pack4(af[g], 0,  p0, s0, h0, bn);
                pack4(af[g], 4,  p1, s1, h1, bn);
                pack4(af[g], 8,  p2, s2, h2, bn);
                pack4(af[g], 12, p3, s3, h3, bn);
            } else {
                #pragma unroll
                for (int j = 0; j < 16; ++j) af[g][j] = (__bf16)0.0f;
            }
        }
        #pragma unroll
        for (int a = 0; a < 4; ++a) {
            const unsigned col = colBase + a * 16 + r;
            bf16x16 bfr;
            #pragma unroll
            for (int j = 0; j < 16; ++j)
                bfr[j] = (__bf16)W[(unsigned)(kb + (j < 8 ? 0 : 8) + hk * 8 + j) * HDIM + col];
            acc[0][a] = __builtin_amdgcn_wmma_f32_16x16x32_bf16(
                false, af[0], false, bfr, (short)0, acc[0][a], false, false);
            acc[1][a] = __builtin_amdgcn_wmma_f32_16x16x32_bf16(
                false, af[1], false, bfr, (short)0, acc[1][a], false, false);
        }
    }

    // epilogue: bias + addv + relu, store, column stats
    #pragma unroll
    for (int g = 0; g < 2; ++g) {
        #pragma unroll
        for (int a = 0; a < 4; ++a) {
            const unsigned col = colBase + a * 16 + r;
            const float bcol = bias[col];
            float s = 0.f, q = 0.f;
            if (blockFull) {
                #pragma unroll
                for (int i = 0; i < 8; ++i) {
                    const unsigned rr = rowBase + g * 16 + i + hk * 8;
                    const unsigned off = rr * HDIM + col;
                    float v = acc[g][a][i] + bcol;
                    if (addv) v += addv[off];
                    v = fmaxf(v, 0.f);
                    out[off] = v;
                    s += v; q += v * v;
                }
            } else {
                #pragma unroll
                for (int i = 0; i < 8; ++i) {
                    const int rr = rowBase + g * 16 + i + hk * 8;
                    if (rr < N) {
                        const unsigned off = (unsigned)rr * HDIM + col;
                        float v = acc[g][a][i] + bcol;
                        if (addv) v += addv[off];
                        v = fmaxf(v, 0.f);
                        out[off] = v;
                        s += v; q += v * v;
                    }
                }
            }
            atomicAdd(&s_sum[col], s);   // ds_add_f32
            atomicAdd(&s_ssq[col], q);
        }
    }
    __syncthreads();
    if (tid < HDIM) {
        atomicAdd(&stats[tid], s_sum[tid]);
        atomicAdd(&stats[HDIM + tid], s_ssq[tid]);
    }
}

// ---------------------------------------------------------- BN finalize -----
__global__ void finalize_kernel(const float* __restrict__ stats,
                                const float* __restrict__ gamma,
                                const float* __restrict__ beta,
                                float* __restrict__ scale,
                                float* __restrict__ shift, float invN) {
    int c = threadIdx.x;
    float mu  = stats[c] * invN;
    float var = stats[HDIM + c] * invN - mu * mu;
    float g   = gamma[c] * rsqrtf(var + BN_EPS);
    scale[c] = g;
    shift[c] = beta[c] - mu * g;
}

// ------------------------------------------------------------- BN apply -----
__global__ __launch_bounds__(256)
void bn_apply_kernel(const float* __restrict__ v,
                     const float* __restrict__ scale,
                     const float* __restrict__ shift,
                     float* __restrict__ outp, long total) {
    long idx = ((long)blockIdx.x * 256 + threadIdx.x) * 4;
    if (idx >= total) return;
    unsigned uidx = (unsigned)idx;
    int c = (int)(uidx & (HDIM - 1));
    float4 val = *(const float4*)(v + uidx);
    float4 sc  = *(const float4*)(scale + c);
    float4 sh  = *(const float4*)(shift + c);
    val.x = val.x * sc.x + sh.x;
    val.y = val.y * sc.y + sh.y;
    val.z = val.z * sc.z + sh.z;
    val.w = val.w * sc.w + sh.w;
    *(float4*)(outp + uidx) = val;
}

// --------------------------------------------------------------- launch -----
extern "C" void kernel_launch(void* const* d_in, const int* in_sizes, int n_in,
                              void* d_out, int out_size, void* d_ws, size_t ws_size,
                              hipStream_t stream) {
    const float* x      = (const float*)d_in[0];
    const float* w      = (const float*)d_in[1];
    const int*   src    = (const int*)  d_in[2];
    const int*   dst    = (const int*)  d_in[3];
    const float* aW     = (const float*)d_in[4];
    const float* ab     = (const float*)d_in[5];
    const float* bW0    = (const float*)d_in[6];
    const float* bb0    = (const float*)d_in[7];
    const float* gamma0 = (const float*)d_in[8];
    const float* beta0  = (const float*)d_in[9];
    const float* bondW  = (const float*)d_in[10];
    const float* bondb  = (const float*)d_in[11];
    const float* W1     = (const float*)d_in[12];
    const float* b1     = (const float*)d_in[13];
    const float* W2     = (const float*)d_in[14];
    const float* b2     = (const float*)d_in[15];
    const float* gamma1 = (const float*)d_in[16];
    const float* beta1  = (const float*)d_in[17];
    const float* gamma2 = (const float*)d_in[18];
    const float* beta2  = (const float*)d_in[19];

    const int A = 128, L = 3;
    const int N = in_sizes[0] / A;       // 50000
    const int E = in_sizes[2];           // 800000
    const long NH = (long)N * HDIM;

    // workspace: acc1 | acc2 | u | stats(512) | scale(256) | shift(256)
    float* acc1   = (float*)d_ws;
    float* acc2   = acc1 + NH;
    float* ubuf   = acc2 + NH;
    float* stats  = ubuf + NH;
    float* scale  = stats + 512;
    float* shiftv = scale + 256;
    float* feat   = (float*)d_out;       // feat lives in d_out across layers

    const float invN = 1.0f / (float)N;
    dim3 blk(256);
    const int edgeGrid = (E + ETILE - 1) / ETILE;
    const int gemmGrid = (N + 63) / 64;
    const int bnGrid   = (int)((NH / 4 + 255) / 256);

    auto zero = [&](float* p, long n) {
        int g = (int)(((n + 3) / 4 + 255) / 256);
        zero_kernel<<<g, blk, 0, stream>>>(p, n);
    };

    // ---- layer 0: feat = BN0(relu(seg(w@bW0+bb0) + x@aW + ab)) ----
    zero(acc2, NH);
    zero(stats, 512);
    edge_kernel<<<edgeGrid, blk, 0, stream>>>(w, bW0, bb0, dst,
                                              nullptr, nullptr, nullptr, acc2, E, 0);
    node_gemm_kernel<<<gemmGrid, blk, 0, stream>>>(x, A, nullptr, nullptr,
                                                   aW, ab, acc2, ubuf, stats, N);
    finalize_kernel<<<1, 256, 0, stream>>>(stats, gamma0, beta0, scale, shiftv, invN);
    bn_apply_kernel<<<bnGrid, blk, 0, stream>>>(ubuf, scale, shiftv, feat, NH);

    // ---- message-passing layers ----
    for (int l = 0; l < L; ++l) {
        zero(acc1, 2 * NH);              // acc1 and acc2 are contiguous
        zero(stats, 512);
        edge_kernel<<<edgeGrid, blk, 0, stream>>>(
            w, bondW + (long)l * BDIM * HDIM, bondb + (long)l * HDIM,
            dst, src, feat, acc1, acc2, E, 1);
        // u = relu(acc1 @ W1 + b1 + acc2)
        node_gemm_kernel<<<gemmGrid, blk, 0, stream>>>(
            acc1, HDIM, nullptr, nullptr,
            W1 + (long)l * HDIM * HDIM, b1 + (long)l * HDIM, acc2, ubuf, stats, N);
        finalize_kernel<<<1, 256, 0, stream>>>(stats, gamma1 + (long)l * HDIM,
                                               beta1 + (long)l * HDIM, scale, shiftv, invN);
        zero(stats, 512);
        // v = relu(BN1(u) @ W2 + b2 + feat)  (BN1 folded into A-load); v -> acc2
        node_gemm_kernel<<<gemmGrid, blk, 0, stream>>>(
            ubuf, HDIM, scale, shiftv,
            W2 + (long)l * HDIM * HDIM, b2 + (long)l * HDIM, feat, acc2, stats, N);
        finalize_kernel<<<1, 256, 0, stream>>>(stats, gamma2 + (long)l * HDIM,
                                               beta2 + (long)l * HDIM, scale, shiftv, invN);
        bn_apply_kernel<<<bnGrid, blk, 0, stream>>>(acc2, scale, shiftv, feat, NH);
    }
    (void)n_in; (void)out_size; (void)ws_size;
}
